// multi_head_attention_46368466927918
// MI455X (gfx1250) — matrix operand order repaired and checked by reading, not compiled, not run on hardware
//
#include <hip/hip_runtime.h>
#include <hip/hip_bf16.h>

// MHA for MI455X (gfx1250):
//  - all matmuls on v_wmma_f32_16x16x32_f16 (wave32), f32 accumulation
//  - block-cooperative causal flash attention: K tile staged to LDS via the
//    Tensor Data Mover (tensor_load_to_lds + s_wait_tensorcnt), V tile via
//    global_load_async_to_lds_b128 (+ s_wait_asynccnt); B-operands read from
//    LDS once per block instead of once per wave.

typedef _Float16 v16h __attribute__((ext_vector_type(16)));
typedef _Float16 v8h  __attribute__((ext_vector_type(8)));
typedef float    v8f  __attribute__((ext_vector_type(8)));
typedef unsigned int u32x4 __attribute__((ext_vector_type(4)));
typedef int          i32x8 __attribute__((ext_vector_type(8)));
typedef int          i32x4 __attribute__((ext_vector_type(4)));

constexpr int Bn = 4, T = 2048, D = 1024, H = 16, DH = 64;
constexpr int M_TOTAL = Bn * T;  // 8192

__device__ inline v8f wmma16(v16h a, v16h b, v8f c) {
  return __builtin_amdgcn_wmma_f32_16x16x32_f16(false, a, false, b, (short)0, c,
                                                false, false);
}

// generic pointer -> 32-bit LDS byte offset (addrspacecast flat->local)
typedef __attribute__((address_space(3))) const void lds_cv;
__device__ inline unsigned lds_off(const void* p) {
  return (unsigned)(uintptr_t)(lds_cv*)p;
}

// TDM: 2D tile load Global -> LDS. dim0/stride0/tile0 in elements (2B each).
__device__ inline void tdm_load_2d_f16(unsigned lds_addr, const void* gptr,
                                       unsigned tdim0, unsigned tdim1,
                                       unsigned tile0, unsigned tile1,
                                       unsigned stride0) {
  const unsigned long long ga = (unsigned long long)(uintptr_t)gptr;
  u32x4 g0;
  g0.x = 1u;                                   // count=1, user D#
  g0.y = lds_addr;                             // lds_addr [63:32]
  g0.z = (unsigned)ga;                         // global_addr lo
  g0.w = (unsigned)(ga >> 32) | (2u << 30);    // global_addr hi | type=2
  i32x8 g1;
  g1[0] = (int)(1u << 16);                     // data_size=1 (2 bytes)
  g1[1] = (int)((tdim0 & 0xffffu) << 16);      // tensor_dim0[15:0] @ bits63:48
  g1[2] = (int)((tdim0 >> 16) | ((tdim1 & 0xffffu) << 16));
  g1[3] = (int)((tdim1 >> 16) | (tile0 << 16));// tile_dim0 @ bits127:112
  g1[4] = (int)tile1;                          // tile_dim1 (tile_dim2=0)
  g1[5] = (int)stride0;                        // tensor_dim0_stride lo
  g1[6] = 0;
  g1[7] = 0;
  i32x4 z4 = {0, 0, 0, 0};
  i32x8 z8 = {0, 0, 0, 0, 0, 0, 0, 0};
  __builtin_amdgcn_tensor_load_to_lds(g0, g1, z4, z4, z8, 0);
}

// async 16B copy Global -> LDS
__device__ inline void async_b128(unsigned lds_addr, const void* gptr) {
  unsigned long long ga = (unsigned long long)(uintptr_t)gptr;
  asm volatile("global_load_async_to_lds_b128 %0, %1, off"
               :: "v"(lds_addr), "v"(ga) : "memory");
}

// ---------------------------------------------------------------------------
// W (K x N f32 row-major) -> WT (N x K f16 row-major)
// ---------------------------------------------------------------------------
__global__ void __launch_bounds__(256) wtrans_kernel(const float* __restrict__ W,
                                                     _Float16* __restrict__ WT) {
  int idx = blockIdx.x * blockDim.x + threadIdx.x;
  int n = idx >> 10;
  int k = idx & (D - 1);
  WT[n * D + k] = (_Float16)W[k * D + n];
}

// ---------------------------------------------------------------------------
// Y = X @ W + b via WMMA (wave tile 16x64, block tile 128x64).
// MODE 0: f16 (B,H,T,DH)   MODE 1: f16 (B,H,DH,T)   MODE 2: f32 (M,D)
// ---------------------------------------------------------------------------
template <bool AHALF, int MODE>
__global__ void __launch_bounds__(256) proj_kernel(const void* __restrict__ Xv,
                                                   const _Float16* __restrict__ WT,
                                                   const float* __restrict__ bias,
                                                   void* __restrict__ Out) {
  const int lane = threadIdx.x & 31;
  const int wave = threadIdx.x >> 5;
  const int l15  = lane & 15;
  const int lhi  = lane >> 4;
  const int m0   = blockIdx.x * 128 + wave * 16;
  const int n0   = blockIdx.y * 64;

  v8f acc[4];
#pragma unroll
  for (int s = 0; s < 4; ++s)
#pragma unroll
    for (int r = 0; r < 8; ++r) acc[s][r] = 0.0f;

  const int arow = m0 + l15;
#pragma unroll 2
  for (int k0 = 0; k0 < D; k0 += 32) {
    // B-operands first so the loads can issue as one clause
    v16h bm[4];
#pragma unroll
    for (int s = 0; s < 4; ++s)
      bm[s] = *(const v16h*)(WT + (size_t)(n0 + s * 16 + l15) * D + k0 + lhi * 16);

    v16h a;
    if constexpr (AHALF) {
      const _Float16* p = (const _Float16*)Xv + (size_t)arow * D + k0 + lhi * 16;
      v8h a0 = *(const v8h*)p;
      v8h a1 = *(const v8h*)(p + 8);
#pragma unroll
      for (int i = 0; i < 8; ++i) { a[i] = a0[i]; a[8 + i] = a1[i]; }
    } else {
      const float* p = (const float*)Xv + (size_t)arow * D + k0 + lhi * 16;
      v8f a0 = *(const v8f*)p;
      v8f a1 = *(const v8f*)(p + 8);
#pragma unroll
      for (int i = 0; i < 8; ++i) { a[i] = (_Float16)a0[i]; a[8 + i] = (_Float16)a1[i]; }
    }
#pragma unroll
    for (int s = 0; s < 4; ++s) acc[s] = wmma16(a, bm[s], acc[s]);
  }

#pragma unroll
  for (int s = 0; s < 4; ++s) {
    const int ng = n0 + s * 16 + l15;
    const float bval = bias[ng];
#pragma unroll
    for (int r = 0; r < 8; ++r) {
      const int mg  = m0 + r + lhi * 8;
      const float v = acc[s][r] + bval;
      if constexpr (MODE == 2) {
        ((float*)Out)[(size_t)mg * D + ng] = v;
      } else {
        const int b = mg >> 11, t = mg & (T - 1);
        const int h = ng >> 6, dh = ng & (DH - 1);
        const _Float16 hv = (_Float16)v;
        if constexpr (MODE == 0)
          ((_Float16*)Out)[(((size_t)(b * H + h)) * T + t) * DH + dh] = hv;
        else
          ((_Float16*)Out)[(((size_t)(b * H + h)) * DH + dh) * T + t] = hv;
      }
    }
  }
}

// ---------------------------------------------------------------------------
// Block-cooperative causal flash attention.
// Qh,Kh: (B,H,T,DH) f16.  VT: (B,H,DH,T) f16.  Ctx: (B,T,D) f16.
// Block = 8 waves = 128 q rows of one (b,h); per 32-key chunk:
//   K tile -> LDS via TDM (wave 0), V tile -> LDS via async b128 (all threads),
//   then WMMA B-operands come from LDS.
// ---------------------------------------------------------------------------
__global__ void __launch_bounds__(256) flash_kernel(const _Float16* __restrict__ Qh,
                                                    const _Float16* __restrict__ Kh,
                                                    const _Float16* __restrict__ VT,
                                                    _Float16* __restrict__ Ctx) {
  __shared__ __align__(16) _Float16 Kt[32][64];      // [key_local][dh]   4KB
  __shared__ __align__(16) _Float16 Vt[64][32];      // [dh][key_local]   4KB
  __shared__ __align__(16) _Float16 plds[8][16][32]; // per-wave P tile   8KB

  const int tid  = threadIdx.x;
  const int lane = tid & 31;
  const int wave = tid >> 5;
  const int l15  = lane & 15;
  const int lhi  = lane >> 4;
  const int bh   = blockIdx.y;
  const int q0b  = blockIdx.x * 128;
  const int q0   = q0b + wave * 16;

  const _Float16* Qp = Qh + (size_t)bh * T * DH;
  const _Float16* Kp = Kh + (size_t)bh * T * DH;
  const _Float16* Vp = VT + (size_t)bh * DH * T;

  // Q A-operands for the two DH k-steps, kept in registers.
  v16h aq[2];
  {
    const _Float16* qrow = Qp + (size_t)(q0 + l15) * DH;
#pragma unroll
    for (int ks = 0; ks < 2; ++ks) {
      v8h a0 = *(const v8h*)(qrow + ks * 32 + lhi * 16);
      v8h a1 = *(const v8h*)(qrow + ks * 32 + 8 + lhi * 16);
#pragma unroll
      for (int i = 0; i < 8; ++i) { aq[ks][i] = a0[i]; aq[ks][8 + i] = a1[i]; }
    }
  }

  float mi[8], li[8];
  v8f o[4];
#pragma unroll
  for (int r = 0; r < 8; ++r) { mi[r] = -1e30f; li[r] = 0.0f; }
#pragma unroll
  for (int s = 0; s < 4; ++s)
#pragma unroll
    for (int r = 0; r < 8; ++r) o[s][r] = 0.0f;

  // V-tile async copy: this thread's slice (one b128)
  const int vdh = tid >> 2;         // 0..63
  const int vgr = (tid & 3) * 8;    // 0,8,16,24
  const unsigned vt_lds = lds_off(&Vt[vdh][vgr]);

  const int cmax = (q0b + 127) >> 5;  // uniform across the block
  for (int c = 0; c <= cmax; ++c) {
    // ---- stage K (TDM) and V (async) tiles into LDS ----
    if (wave == 0)
      tdm_load_2d_f16(lds_off(&Kt[0][0]), Kp + (size_t)(c * 32) * DH,
                      /*tdim0=*/DH, /*tdim1=*/32, /*tile0=*/DH, /*tile1=*/32,
                      /*stride0=*/DH);
    async_b128(vt_lds, Vp + (size_t)vdh * T + c * 32 + vgr);
    asm volatile("s_wait_asynccnt 0" ::: "memory");
    if (wave == 0) __builtin_amdgcn_s_wait_tensorcnt(0);
    __syncthreads();

    // ---- S = Q @ K^T (two 16-wide N-subtiles, K-dim = DH = 64) ----
    v8f s0, s1;
#pragma unroll
    for (int r = 0; r < 8; ++r) { s0[r] = 0.0f; s1[r] = 0.0f; }
#pragma unroll
    for (int ks = 0; ks < 2; ++ks) {
      v16h b0 = *(const v16h*)&Kt[l15][ks * 32 + lhi * 16];
      v16h b1 = *(const v16h*)&Kt[16 + l15][ks * 32 + lhi * 16];
      s0 = wmma16(aq[ks], b0, s0);
      s1 = wmma16(aq[ks], b1, s1);
    }

    // ---- scale + causal mask + row max (16-lane shfl_xor reductions) ----
    float p0a[8], p1a[8], mx[8];
#pragma unroll
    for (int r = 0; r < 8; ++r) {
      const int qrow = q0 + r + lhi * 8;
      const int key0 = c * 32 + l15;
      const int key1 = key0 + 16;
      float v0 = s0[r] * 0.125f;
      float v1 = s1[r] * 0.125f;
      if (key0 > qrow) v0 = -1e30f;
      if (key1 > qrow) v1 = -1e30f;
      p0a[r] = v0; p1a[r] = v1;
      float m = fmaxf(v0, v1);
      m = fmaxf(m, __shfl_xor(m, 1, 32));
      m = fmaxf(m, __shfl_xor(m, 2, 32));
      m = fmaxf(m, __shfl_xor(m, 4, 32));
      m = fmaxf(m, __shfl_xor(m, 8, 32));
      mx[r] = m;
    }

    // ---- online softmax update; write P (f16) to per-wave LDS tile ----
#pragma unroll
    for (int r = 0; r < 8; ++r) {
      const float mnew  = fmaxf(mi[r], mx[r]);
      const float alpha = __expf(mi[r] - mnew);
      const float p0 = __expf(p0a[r] - mnew);
      const float p1 = __expf(p1a[r] - mnew);
      float rs = p0 + p1;
      rs += __shfl_xor(rs, 1, 32);
      rs += __shfl_xor(rs, 2, 32);
      rs += __shfl_xor(rs, 4, 32);
      rs += __shfl_xor(rs, 8, 32);
      li[r] = li[r] * alpha + rs;
      mi[r] = mnew;
      const int mrow = r + lhi * 8;
      plds[wave][mrow][l15]      = (_Float16)p0;
      plds[wave][mrow][16 + l15] = (_Float16)p1;
#pragma unroll
      for (int s = 0; s < 4; ++s) o[s][r] *= alpha;
    }

    asm volatile("s_wait_dscnt 0" ::: "memory");

    // ---- P in A-operand layout; O += P @ V (B-operand from LDS V tile) ----
    v16h ap;
    {
      const _Float16* pr = &plds[wave][l15][0];
      v8h a0 = *(const v8h*)(pr + lhi * 16);
      v8h a1 = *(const v8h*)(pr + 8 + lhi * 16);
#pragma unroll
      for (int i = 0; i < 8; ++i) { ap[i] = a0[i]; ap[8 + i] = a1[i]; }
    }
#pragma unroll
    for (int s = 0; s < 4; ++s) {
      v16h bv = *(const v16h*)&Vt[s * 16 + l15][lhi * 16];
      o[s] = wmma16(ap, bv, o[s]);
    }

    __syncthreads();  // protect Kt/Vt before next chunk overwrites them
  }

  // ---- normalize and store ctx (B,T,D) f16 ----
  const int b = bh >> 4, h = bh & (H - 1);
#pragma unroll
  for (int r = 0; r < 8; ++r) {
    const float inv = 1.0f / li[r];
    const int t = q0 + r + lhi * 8;
    const size_t base = ((size_t)(b * T + t)) * D + h * DH;
#pragma unroll
    for (int s = 0; s < 4; ++s)
      Ctx[base + s * 16 + l15] = (_Float16)(o[s][r] * inv);
  }
}

// ---------------------------------------------------------------------------
extern "C" void kernel_launch(void* const* d_in, const int* in_sizes, int n_in,
                              void* d_out, int out_size, void* d_ws, size_t ws_size,
                              hipStream_t stream) {
  const float* q  = (const float*)d_in[0];
  const float* k  = (const float*)d_in[1];
  const float* v  = (const float*)d_in[2];
  const float* Wq = (const float*)d_in[3];
  const float* bq = (const float*)d_in[4];
  const float* Wk = (const float*)d_in[5];
  const float* bk = (const float*)d_in[6];
  const float* Wv = (const float*)d_in[7];
  const float* bv = (const float*)d_in[8];
  const float* Wo = (const float*)d_in[9];
  const float* bo = (const float*)d_in[10];
  float* out = (float*)d_out;

  char* ws = (char*)d_ws;
  size_t off = 0;
  auto carve = [&](size_t bytes) -> void* {
    void* p = ws + off;
    off += (bytes + 255) & ~(size_t)255;
    return p;
  };
  _Float16* wtq = (_Float16*)carve((size_t)D * D * 2);
  _Float16* wtk = (_Float16*)carve((size_t)D * D * 2);
  _Float16* wtv = (_Float16*)carve((size_t)D * D * 2);
  _Float16* wto = (_Float16*)carve((size_t)D * D * 2);
  _Float16* qh  = (_Float16*)carve((size_t)M_TOTAL * D * 2);
  _Float16* kh  = (_Float16*)carve((size_t)M_TOTAL * D * 2);
  _Float16* vt  = (_Float16*)carve((size_t)M_TOTAL * D * 2);
  _Float16* ctx = (_Float16*)carve((size_t)M_TOTAL * D * 2);

  const int wtb = (D * D) / 256;
  wtrans_kernel<<<wtb, 256, 0, stream>>>(Wq, wtq);
  wtrans_kernel<<<wtb, 256, 0, stream>>>(Wk, wtk);
  wtrans_kernel<<<wtb, 256, 0, stream>>>(Wv, wtv);
  wtrans_kernel<<<wtb, 256, 0, stream>>>(Wo, wto);

  dim3 pg(M_TOTAL / 128, D / 64);
  proj_kernel<false, 0><<<pg, 256, 0, stream>>>((const void*)q, wtq, bq, (void*)qh);
  proj_kernel<false, 0><<<pg, 256, 0, stream>>>((const void*)k, wtk, bk, (void*)kh);
  proj_kernel<false, 1><<<pg, 256, 0, stream>>>((const void*)v, wtv, bv, (void*)vt);

  dim3 fg(T / 128, Bn * H);
  flash_kernel<<<fg, 256, 0, stream>>>(qh, kh, vt, ctx);

  proj_kernel<true, 2><<<pg, 256, 0, stream>>>((const void*)ctx, wto, bo, (void*)out);
}
